// GRUDModel_44109314130687
// MI455X (gfx1250) — matrix operand order repaired and checked by reading, not compiled, not run on hardware
//
#include <hip/hip_runtime.h>
#include <hip/hip_bf16.h>

// ---------------------------------------------------------------------------
// GRU-D on MI455X (gfx1250): bf16 WMMA for all GEMMs.
//   B=256, T=512, F=H=256, C=2
// Phase 1 (parallel):
//   pack weights -> bf16 [N][K] transposed layouts
//   pack activations -> bf16 A matrices (x_rep|m concat, deltas)
//   WMMA GEMM: gammaH = exp(-relu(deltas @ W_gh + b_gh))      [T,B,256] f32
//   WMMA GEMM: pre    = [x_rep,m] @ [K_x;K_m] + gru_bias[0]   [T,B,768] f32
// Phase 2 (sequential scan, 16 blocks x 16 batch rows, T=512 steps):
//   h_dec = h * gammaH[t];  acc = h_dec @ [K_h | gru_rec]  (16x256x1536 WMMA)
//   gates -> h_new  (per-wave gate-column ownership, h in LDS)
//   final: softmax(h @ dense_W + dense_b)
// ---------------------------------------------------------------------------

#define BB   256   // batch
#define TT   512   // time steps
#define FF   256   // features == hidden
#define HH   256
#define KPRE 512   // 2F
#define NPRE 768   // 3H
#define NREC 1536  // 3H (K_h) + 3H (gru_rec)

typedef __attribute__((ext_vector_type(8)))  float  v8f;
typedef __attribute__((ext_vector_type(16))) __bf16 v16bf;
typedef __attribute__((ext_vector_type(8)))  __bf16 v8bf;

union AFrag { v16bf v; v8bf h[2]; };

__device__ __forceinline__ __bf16 to_bf16(float f) { return (__bf16)f; }

// --------------------------- weight packing --------------------------------
// WghT [256][256]  : WghT[n][k] = W_gh[k][n]              (gamma_h GEMM B)
// BpreT[768][512]  : k<256 -> gru_kernel[k][n] (K_x) ; k>=256 -> gk[k+256][n] (K_m)
// WrT  [1536][256] : n<768 -> gru_kernel[256+k][n] (K_h) ; n>=768 -> gru_rec[k][n-768]
__global__ void pack_weights(const float* __restrict__ Wgh,
                             const float* __restrict__ gk,
                             const float* __restrict__ gr,
                             __bf16* __restrict__ WghT,
                             __bf16* __restrict__ BpreT,
                             __bf16* __restrict__ WrT) {
  const int stride = gridDim.x * blockDim.x;
  int tid = blockIdx.x * blockDim.x + threadIdx.x;
  for (int i = tid; i < 256 * 256; i += stride) {
    int n = i >> 8, k = i & 255;
    WghT[i] = to_bf16(Wgh[k * HH + n]);
  }
  for (int i = tid; i < NPRE * KPRE; i += stride) {
    int n = i / KPRE, k = i % KPRE;
    int row = (k < 256) ? k : (k + 256);
    BpreT[i] = to_bf16(gk[row * NPRE + n]);
  }
  for (int i = tid; i < NREC * 256; i += stride) {
    int n = i >> 8, k = i & 255;
    WrT[i] = to_bf16((n < NPRE) ? gk[(256 + k) * NPRE + n]
                                : gr[k * NPRE + (n - NPRE)]);
  }
}

// --------------------------- activation packing ----------------------------
// gamma_x, x_rep elementwise; emit bf16 A matrices with M index = t*B + b.
__global__ void pack_act(const float* __restrict__ X,
                         const float* __restrict__ Xl,
                         const float* __restrict__ Mm,
                         const float* __restrict__ D,
                         const float* __restrict__ emp,
                         const float* __restrict__ Wgx,
                         const float* __restrict__ bgx,
                         __bf16* __restrict__ Agh,
                         __bf16* __restrict__ Apre) {
  const long long total = (long long)BB * TT * FF;
  const long long stride = (long long)gridDim.x * blockDim.x;
  for (long long i = (long long)blockIdx.x * blockDim.x + threadIdx.x;
       i < total; i += stride) {
    int f = (int)(i % FF);
    long long bt = i / FF;
    int t = (int)(bt % TT);
    long long b = bt / TT;
    float d  = D[i];
    float m  = Mm[i];
    float x  = X[i];
    float xl = Xl[i];
    float gx = __expf(-fmaxf(d * Wgx[f * FF + f] + bgx[f], 0.f));
    float xh = gx * xl + (1.f - gx) * emp[f];
    float xr = m * x + (1.f - m) * xh;
    long long mrow = (long long)t * BB + b;
    Agh [mrow * FF + f]            = to_bf16(d);
    Apre[mrow * KPRE + f]          = to_bf16(xr);
    Apre[mrow * KPRE + FF + f]     = to_bf16(m);
  }
}

// --------------------------- generic WMMA GEMM -----------------------------
// C[M][N] = epilogue(A[M][K](bf16) @ BT[N][K](bf16) + bias[N])
// mode 0: identity   mode 1: exp(-relu(.))
// block = 256 thr (8 waves), tile 128(M) x 64(N); wave w -> Mtile w, 4 N-tiles
__global__ void __launch_bounds__(256)
wmma_gemm(const __bf16* __restrict__ A, const __bf16* __restrict__ BT,
          const float* __restrict__ bias, float* __restrict__ C,
          int M, int N, int K, int mode) {
  const int w    = threadIdx.x >> 5;
  const int lane = threadIdx.x & 31;
  const int nl   = lane & 15;
  const int half = lane >> 4;
  const int Mbase = blockIdx.y * 128 + w * 16;
  const int Nblk  = blockIdx.x * 64;

  v8f acc[4] = {};
  const __bf16* arowBase = A + (size_t)(Mbase + nl) * K + half * 16;
  for (int kb = 0; kb < K; kb += 32) {
    AFrag af;
    af.h[0] = *(const v8bf*)(arowBase + kb);
    af.h[1] = *(const v8bf*)(arowBase + kb + 8);
#pragma unroll
    for (int j = 0; j < 4; ++j) {
      const __bf16* brow = BT + (size_t)(Nblk + j * 16 + nl) * K + kb + half * 16;
      v16bf bv = *(const v16bf*)brow;
      acc[j] = __builtin_amdgcn_wmma_f32_16x16x32_bf16(
          false, af.v, false, bv, (short)0, acc[j], false, false);
    }
  }
#pragma unroll
  for (int j = 0; j < 4; ++j) {
    int n = Nblk + j * 16 + nl;
    float bv = bias[n];
#pragma unroll
    for (int v = 0; v < 8; ++v) {
      int m = Mbase + v + 8 * half;
      float val = acc[j][v] + bv;
      if (mode == 1) val = __expf(-fmaxf(val, 0.f));
      C[(size_t)m * N + n] = val;
    }
  }
}

// --------------------------- sequential GRU-D scan -------------------------
// 16 blocks x 16 batch rows; h lives in LDS; per step one 16x256x1536 bf16
// WMMA GEMM; wave w owns gate columns [32w, 32w+32) across all 6 segments.
__global__ void __launch_bounds__(256)
grud_scan(const float* __restrict__ pre,     // [T][B][768]
          const float* __restrict__ gammaH,  // [T][B][256]
          const __bf16* __restrict__ WrT,    // [1536][256]
          const float* __restrict__ gru_bias,// [2][768]
          const float* __restrict__ dW,      // [256][2]
          const float* __restrict__ db,      // [2]
          float* __restrict__ out) {         // [B][2]
  __shared__ __align__(16) float  hnew[16][256];
  __shared__ __align__(16) float  hdec[16][256];
  __shared__ __align__(16) __bf16 hbf [16][256];

  const int tid  = threadIdx.x;
  const int w    = tid >> 5;
  const int lane = tid & 31;
  const int nl   = lane & 15;
  const int half = lane >> 4;
  const int Bbase = blockIdx.x * 16;
  const float* gb1 = gru_bias + NPRE;

  for (int r = 0; r < 16; ++r) hnew[r][tid] = 0.f;
  __syncthreads();

  for (int t = 0; t < TT; ++t) {
    // (a) decay previous hidden state; publish f32 + bf16 copies
    const float* gH = gammaH + ((size_t)t * BB + Bbase) * HH;
    for (int r = 0; r < 16; ++r) {
      float hd = hnew[r][tid] * gH[(size_t)r * HH + tid];
      hdec[r][tid] = hd;
      hbf [r][tid] = to_bf16(hd);
    }
    __syncthreads();

    // (b) 16x256 @ 256x1536 bf16 WMMA GEMM (12 N-tiles per wave)
    v8f acc[12] = {};
#pragma unroll
    for (int kt = 0; kt < 8; ++kt) {
      AFrag af;
      af.h[0] = *(const v8bf*)&hbf[nl][kt * 32 + half * 16];
      af.h[1] = *(const v8bf*)&hbf[nl][kt * 32 + half * 16 + 8];
#pragma unroll
      for (int s = 0; s < 6; ++s) {
#pragma unroll
        for (int j = 0; j < 2; ++j) {
          int n = s * 256 + w * 32 + j * 16 + nl;
          v16bf bv = *(const v16bf*)(WrT + (size_t)n * HH + kt * 32 + half * 16);
          acc[s * 2 + j] = __builtin_amdgcn_wmma_f32_16x16x32_bf16(
              false, af.v, false, bv, (short)0, acc[s * 2 + j], false, false);
        }
      }
    }

    // (c) GRU gates; wave w owns hidden columns [32w, 32w+32)
    const float* preT = pre + ((size_t)t * BB + Bbase) * NPRE;
#pragma unroll
    for (int j = 0; j < 2; ++j) {
      int n = w * 32 + j * 16 + nl;
#pragma unroll
      for (int v = 0; v < 8; ++v) {
        int m = v + 8 * half;
        const float* pr = preT + (size_t)m * NPRE;
        float xz = pr[n]       + acc[0 + j][v];
        float xr = pr[256 + n] + acc[2 + j][v];
        float xc = pr[512 + n] + acc[4 + j][v];
        float rz = gb1[n]       + acc[6 + j][v];
        float rr = gb1[256 + n] + acc[8 + j][v];
        float rc = gb1[512 + n] + acc[10 + j][v];
        float z  = 1.f / (1.f + __expf(-(xz + rz)));
        float r  = 1.f / (1.f + __expf(-(xr + rr)));
        float hh = tanhf(xc + r * rc);
        float ho = hdec[m][n];
        hnew[m][n] = z * ho + (1.f - z) * hh;
      }
    }
    __syncthreads();
  }

  // classifier + softmax (C=2)
  if (tid < 16) {
    float a0 = db[0], a1 = db[1];
    for (int k = 0; k < 256; ++k) {
      float hv = hnew[tid][k];
      a0 += hv * dW[k * 2 + 0];
      a1 += hv * dW[k * 2 + 1];
    }
    float mx = fmaxf(a0, a1);
    float e0 = __expf(a0 - mx), e1 = __expf(a1 - mx);
    float s = e0 + e1;
    out[(Bbase + tid) * 2 + 0] = e0 / s;
    out[(Bbase + tid) * 2 + 1] = e1 / s;
  }
}

// ---------------------------------------------------------------------------
extern "C" void kernel_launch(void* const* d_in, const int* in_sizes, int n_in,
                              void* d_out, int out_size, void* d_ws, size_t ws_size,
                              hipStream_t stream) {
  const float* X    = (const float*)d_in[0];
  const float* Xl   = (const float*)d_in[1];
  const float* Mm   = (const float*)d_in[2];
  const float* D    = (const float*)d_in[3];
  const float* emp  = (const float*)d_in[4];
  const float* Wgh  = (const float*)d_in[5];
  const float* bgh  = (const float*)d_in[6];
  const float* Wgx  = (const float*)d_in[7];
  const float* bgx  = (const float*)d_in[8];
  const float* gk   = (const float*)d_in[9];
  const float* gr   = (const float*)d_in[10];
  const float* gb   = (const float*)d_in[11];
  const float* dW   = (const float*)d_in[12];
  const float* db   = (const float*)d_in[13];
  float* out = (float*)d_out;

  // workspace carve-up (256B aligned)
  char* ws = (char*)d_ws;
  size_t off = 0;
  auto carve = [&](size_t bytes) -> char* {
    char* p = ws + off;
    off = (off + bytes + 255) & ~(size_t)255;
    return p;
  };
  const size_t MROWS = (size_t)BB * TT;
  __bf16* Agh   = (__bf16*)carve(MROWS * FF * sizeof(__bf16));
  __bf16* Apre  = (__bf16*)carve(MROWS * KPRE * sizeof(__bf16));
  __bf16* WghT  = (__bf16*)carve((size_t)HH * FF * sizeof(__bf16));
  __bf16* BpreT = (__bf16*)carve((size_t)NPRE * KPRE * sizeof(__bf16));
  __bf16* WrT   = (__bf16*)carve((size_t)NREC * HH * sizeof(__bf16));
  float*  gammaH = (float*)carve(MROWS * HH * sizeof(float));
  float*  pre    = (float*)carve(MROWS * NPRE * sizeof(float));
  (void)ws_size; (void)in_sizes; (void)n_in; (void)out_size;

  pack_weights<<<1024, 256, 0, stream>>>(Wgh, gk, gr, WghT, BpreT, WrT);
  pack_act<<<8192, 256, 0, stream>>>(X, Xl, Mm, D, emp, Wgx, bgx, Agh, Apre);

  // gammaH = exp(-relu(deltas @ W_gh + b_gh)) : M=131072, N=256, K=256
  wmma_gemm<<<dim3(HH / 64, MROWS / 128), 256, 0, stream>>>(
      Agh, WghT, bgh, gammaH, (int)MROWS, HH, FF, /*mode=*/1);

  // pre = [x_rep,m] @ [K_x;K_m] + gru_bias[0] : M=131072, N=768, K=512
  wmma_gemm<<<dim3(NPRE / 64, MROWS / 128), 256, 0, stream>>>(
      Apre, BpreT, gb, pre, (int)MROWS, NPRE, KPRE, /*mode=*/0);

  grud_scan<<<BB / 16, 256, 0, stream>>>(pre, gammaH, WrT, gb, dW, db, out);
}
